// SelfAttention_65481071395953
// MI455X (gfx1250) — compile-verified, matrix operand order checked by reading, not run on hardware
//
#include <hip/hip_runtime.h>
#include <hip/hip_bf16.h>

// ---------------------------------------------------------------------------
// Self-attention (SAGAN-style) for B=8, C=512, H=W=64 (N=4096), D=C/16=32.
//   q = Wq@x, k = Wk@x, v = Wv@x           (1x1 conv == GEMM over channels)
//   S = q^T k  [N x N];  P[m,n] = exp(S[m,n]-rowmax[m]) / rowsum[m]
//   out = v @ P                            (contract over m)
// Flash-style: P never touches global memory. All heavy math is
// v_wmma_f32_16x16x32_f16 (wave32). V is stored in WMMA A-operand tile order
// so the dominant GEMM's loads are single contiguous 32B-per-lane b128s.
// ---------------------------------------------------------------------------

typedef __attribute__((ext_vector_type(16))) _Float16 v16h;
typedef __attribute__((ext_vector_type(8)))  _Float16 v8h;
typedef __attribute__((ext_vector_type(8)))  float    v8f;
typedef __attribute__((ext_vector_type(4)))  float    v4f;

constexpr int BN = 8;     // batches
constexpr int CC = 512;   // channels
constexpr int NN = 4096;  // spatial positions (64*64)
constexpr int DD = 32;    // q/k channels
constexpr int MS = NN / 32;  // 32-wide m-steps per row

// WMMA f16 16x16x32 operand layout (A and B symmetric):
//   lane L: owns row/col (L&15); koff = (L>=16)?8:0
//   v16h elements i<8  -> K = koff + i ; i>=8 -> K = koff + 16 + (i-8)
// C/D layout: lane L holds column N=(L&15); VGPR j holds row M = j + (L>=16?8:0).

__device__ __forceinline__ v16h load_op16(const _Float16* p) {
  const v8h* p8 = (const v8h*)p;
  v8h lo = p8[0];          // K = koff + 0..7
  v8h hi = p8[2];          // K = koff + 16..23
  return __builtin_shufflevector(lo, hi, 0,1,2,3,4,5,6,7,8,9,10,11,12,13,14,15);
}

__device__ __forceinline__ v8f wmma_f16(v16h a, v16h b, v8f c) {
  return __builtin_amdgcn_wmma_f32_16x16x32_f16(false, a, false, b,
                                                (short)0, c, false, false);
}

// ---------------------------------------------------------------------------
// Kernel 1: projection GEMM  out[d,n] = sum_c W[d,c] * x[b,c,n], f16 output.
// Each wave computes ROWT row-tiles for ONE 16-wide n-tile, so the expensive
// strided x B-tile is loaded once per K-step and reused by ROWT WMMAs.
// grid = (NN/16, rows/(16*ROWT), BN), block = 32.
// MODE 1: store out as [n][d]           (Q, K — feeds A/B operands of S)
// MODE 0: store out in A-operand tiles  (V — feeds A operand of out-GEMM)
//         tile (ct, ms) = 16c x 32m, 512 f16; element (c,m) at
//         lane' = (c&15) + 16*((mrel>>3)&1), i' = (mrel&7) + ((mrel>>4)&1)*8.
// ---------------------------------------------------------------------------
template <int ROWT, int MODE>
__global__ __launch_bounds__(32) void proj_kernel(
    const float* __restrict__ Wt, const float* __restrict__ x,
    _Float16* __restrict__ out, int rows) {
  const int lane = threadIdx.x & 31;
  const int half = lane >> 4;
  const int l15  = lane & 15;
  const int koff = half * 8;
  const int n0 = blockIdx.x * 16;
  const int d0 = blockIdx.y * 16 * ROWT;
  const int b  = blockIdx.z;

  const float* xb = x + (size_t)b * CC * NN;
  v8f acc[ROWT] = {};
  for (int c0 = 0; c0 < CC; c0 += 32) {
    // B tile: x[c0 + k][n0 + l15]  (stride-N loads, coalesced across lanes)
    const float* xcol = xb + (size_t)(c0 + koff) * NN + n0 + l15;
    v16h bm;
#pragma unroll
    for (int i = 0; i < 8; ++i) bm[i]     = (_Float16)xcol[(size_t)i * NN];
#pragma unroll
    for (int i = 0; i < 8; ++i) bm[8 + i] = (_Float16)xcol[(size_t)(16 + i) * NN];

#pragma unroll
    for (int r = 0; r < ROWT; ++r) {
      // A tile: W[d0+16r+l15][c0 + koff + {0..7,16..23}] (W is L2-resident)
      const float* wrow = Wt + (size_t)(d0 + 16 * r + l15) * CC + c0 + koff;
      const v4f* w4 = (const v4f*)wrow;
      v4f w0 = w4[0], w1 = w4[1], w2 = w4[4], w3 = w4[5];
      v16h a;
#pragma unroll
      for (int i = 0; i < 4; ++i) {
        a[i]      = (_Float16)w0[i];
        a[4 + i]  = (_Float16)w1[i];
        a[8 + i]  = (_Float16)w2[i];
        a[12 + i] = (_Float16)w3[i];
      }
      acc[r] = wmma_f16(a, bm, acc[r]);
    }
  }

  if (MODE == 1) {
#pragma unroll
    for (int r = 0; r < ROWT; ++r) {
      v8h o;
#pragma unroll
      for (int j = 0; j < 8; ++j) o[j] = (_Float16)acc[r][j];
      _Float16* dst = out + (size_t)b * NN * rows +
                      (size_t)(n0 + l15) * rows + d0 + 16 * r + half * 8;
      *(v8h*)dst = o;
    }
  } else {
    // A-operand tiled store. This lane's column is m = n0 + l15 (uniform ms).
    const int ms   = n0 >> 5;
    const int mrel = (n0 & 16) + l15;
    const int bsel = (mrel >> 3) & 1;
    const int ip   = (mrel & 7) + ((mrel >> 4) & 1) * 8;
    _Float16* dstb = out + (size_t)b * CC * NN + ip;
#pragma unroll
    for (int r = 0; r < ROWT; ++r) {
      _Float16* tbase = dstb + ((size_t)(d0 / 16 + r) * MS + ms) * 512;
#pragma unroll
      for (int j = 0; j < 8; ++j)   // rows c = d0+16r + j + half*8
        tbase[(j + half * 8 + 16 * bsel) * 16] = (_Float16)acc[r][j];
    }
  }
}

// ---------------------------------------------------------------------------
// Kernel 2: softmax row statistics over n for rows m of S = q^T k.
// Each wave handles MT=4 m-tiles: every K B-tile is loaded once and reused by
// 4 WMMAs. grid = (NN/64, BN), block = 32. Writes rowmax[m] and 1/rowsum[m].
// ---------------------------------------------------------------------------
__global__ __launch_bounds__(32) void stats_kernel(
    const _Float16* __restrict__ QT, const _Float16* __restrict__ KT,
    float* __restrict__ rowmax, float* __restrict__ rowrcp) {
  constexpr int MT = 4;
  const int lane = threadIdx.x & 31;
  const int half = lane >> 4;
  const int l15  = lane & 15;
  const int koff = half * 8;
  const int m0 = blockIdx.x * 16 * MT;
  const int b  = blockIdx.y;

  const _Float16* QTb = QT + (size_t)b * NN * DD;
  const _Float16* KTb = KT + (size_t)b * NN * DD;

  v16h aq[MT];
#pragma unroll
  for (int t = 0; t < MT; ++t)
    aq[t] = load_op16(QTb + (size_t)(m0 + 16 * t + l15) * DD + koff);

  float rmax[MT][8];
#pragma unroll
  for (int t = 0; t < MT; ++t)
#pragma unroll
    for (int j = 0; j < 8; ++j) rmax[t][j] = -3.4e38f;

  for (int n0 = 0; n0 < NN; n0 += 16) {
    v16h bk = load_op16(KTb + (size_t)(n0 + l15) * DD + koff);
#pragma unroll
    for (int t = 0; t < MT; ++t) {
      v8f z = {};
      v8f s = wmma_f16(aq[t], bk, z);
#pragma unroll
      for (int j = 0; j < 8; ++j) rmax[t][j] = fmaxf(rmax[t][j], s[j]);
    }
  }
#pragma unroll
  for (int mask = 1; mask < 16; mask <<= 1)
#pragma unroll
    for (int t = 0; t < MT; ++t)
#pragma unroll
      for (int j = 0; j < 8; ++j)
        rmax[t][j] = fmaxf(rmax[t][j], __shfl_xor(rmax[t][j], mask, 32));

  float rsum[MT][8];
#pragma unroll
  for (int t = 0; t < MT; ++t)
#pragma unroll
    for (int j = 0; j < 8; ++j) rsum[t][j] = 0.0f;

  for (int n0 = 0; n0 < NN; n0 += 16) {
    v16h bk = load_op16(KTb + (size_t)(n0 + l15) * DD + koff);
#pragma unroll
    for (int t = 0; t < MT; ++t) {
      v8f z = {};
      v8f s = wmma_f16(aq[t], bk, z);
#pragma unroll
      for (int j = 0; j < 8; ++j) rsum[t][j] += __expf(s[j] - rmax[t][j]);
    }
  }
#pragma unroll
  for (int mask = 1; mask < 16; mask <<= 1)
#pragma unroll
    for (int t = 0; t < MT; ++t)
#pragma unroll
      for (int j = 0; j < 8; ++j) rsum[t][j] += __shfl_xor(rsum[t][j], mask, 32);

  if (l15 == 0) {  // lanes 0 and 16 write rows 0..7 and 8..15 of each tile
#pragma unroll
    for (int t = 0; t < MT; ++t) {
      float* dm = rowmax + (size_t)b * NN + m0 + 16 * t + half * 8;
      float* dr = rowrcp + (size_t)b * NN + m0 + 16 * t + half * 8;
#pragma unroll
      for (int j = 0; j < 8; ++j) { dm[j] = rmax[t][j]; dr[j] = 1.0f / rsum[t][j]; }
    }
  }
}

// ---------------------------------------------------------------------------
// Kernel 3: out[:, n_blk] = V @ P, P generated on the fly per 128-row m chunk.
// Workgroup = 16 waves (512 threads), owns a 512c x 64n output block:
//   wave w: c-tiles {32w, 32w+16}; accumulators for all 4 n-subtiles.
// Producer: wave w computes S tiles (m-tile = w&7, n-subtiles (w>>3)*2+{0,1});
// the WMMA D-register layout IS the consumer WMMA's B-operand layout, so LDS
// staging is shuffle-free. V A-tiles are single 32B-per-lane b128 loads from
// the tiled layout; each feeds 4 WMMAs. grid = (NN/64, BN), block = 512.
// ---------------------------------------------------------------------------
__global__ __launch_bounds__(512) void attn_out_kernel(
    const _Float16* __restrict__ QT, const _Float16* __restrict__ KT,
    const _Float16* __restrict__ Vt, const float* __restrict__ rowmax,
    const float* __restrict__ rowrcp, float* __restrict__ out) {
  const int tid  = threadIdx.x;
  const int wave = tid >> 5;          // 0..15
  const int lane = tid & 31;
  const int half = lane >> 4;
  const int l15  = lane & 15;
  const int koff = half * 8;
  const int n0 = blockIdx.x * 64;
  const int b  = blockIdx.y;
  const int mt  = wave & 7;           // produced m-tile within chunk
  const int ntp = (wave >> 3) * 2;    // first produced n-subtile

  const _Float16* QTb = QT + (size_t)b * NN * DD;
  const _Float16* KTb = KT + (size_t)b * NN * DD;
  const _Float16* Vtb = Vt + (size_t)b * CC * NN;  // A-operand tiled layout
  const float* rmaxb = rowmax + (size_t)b * NN;
  const float* rrcpb = rowrcp + (size_t)b * NN;

  // K tiles (B operand of S) for this wave's two produced n-subtiles.
  v16h bkp[2];
#pragma unroll
  for (int u = 0; u < 2; ++u)
    bkp[u] = load_op16(KTb + (size_t)(n0 + 16 * (ntp + u) + l15) * DD + koff);

  __shared__ _Float16 pbuf[32 * 256];  // (8 m-tiles x 4 n-tiles) x 512B = 16 KB

  v8f acc[8] = {};                     // [c-tile t(2)][n-subtile nt(4)]
  for (int mc = 0; mc < NN; mc += 128) {
    // ---- produce: two 16x16 S tiles -> P (f16) -> LDS ----
    const int m0p = mc + mt * 16;
    v16h aq = load_op16(QTb + (size_t)(m0p + l15) * DD + koff);
    const v4f* rm4 = (const v4f*)(rmaxb + m0p + half * 8);
    const v4f* rr4 = (const v4f*)(rrcpb + m0p + half * 8);
    v4f rm0 = rm4[0], rm1 = rm4[1], rr0 = rr4[0], rr1 = rr4[1];
    float rmv[8], rrv[8];
#pragma unroll
    for (int j = 0; j < 4; ++j) {
      rmv[j] = rm0[j]; rmv[4 + j] = rm1[j];
      rrv[j] = rr0[j]; rrv[4 + j] = rr1[j];
    }

    v8h p[2];
#pragma unroll
    for (int u = 0; u < 2; ++u) {
      v8f z = {};
      v8f s = wmma_f16(aq, bkp[u], z);
#pragma unroll
      for (int j = 0; j < 8; ++j)
        p[u][j] = (_Float16)(__expf(s[j] - rmv[j]) * rrv[j]);
    }

    __syncthreads();  // previous chunk fully consumed
#pragma unroll
    for (int u = 0; u < 2; ++u)
      *(v8h*)&pbuf[(mt * 4 + ntp + u) * 256 + lane * 8] = p[u];
    __syncthreads();  // P chunk visible to all waves

    // Prefetch next chunk's V tiles (global_prefetch_b8; lanes cover 1KB/tile).
    if (mc + 128 < NN) {
      const int msn = (mc + 128) >> 5;
#pragma unroll
      for (int t = 0; t < 2; ++t)
#pragma unroll
        for (int s4 = 0; s4 < 4; ++s4)
          __builtin_prefetch(
              Vtb + ((size_t)(wave * 2 + t) * MS + msn + s4) * 512 + lane * 16,
              0, 1);
    }

    // ---- consume: 4 m-steps of 32; each V A-tile feeds 4 WMMAs ----
#pragma unroll
    for (int s4 = 0; s4 < 4; ++s4) {
      v16h bp[4];
#pragma unroll
      for (int nt = 0; nt < 4; ++nt) {
        v8h blo = *(const v8h*)&pbuf[((2 * s4) * 4 + nt) * 256 + lane * 8];
        v8h bhi = *(const v8h*)&pbuf[((2 * s4 + 1) * 4 + nt) * 256 + lane * 8];
        bp[nt] = __builtin_shufflevector(blo, bhi,
                 0,1,2,3,4,5,6,7,8,9,10,11,12,13,14,15);
      }
      const int ms = (mc >> 5) + s4;
#pragma unroll
      for (int t = 0; t < 2; ++t) {
        const int ct = wave * 2 + t;  // c-tile index
        v16h av = *(const v16h*)(Vtb + ((size_t)ct * MS + ms) * 512 + lane * 16);
#pragma unroll
        for (int nt = 0; nt < 4; ++nt)
          acc[t * 4 + nt] = wmma_f16(av, bp[nt], acc[t * 4 + nt]);
      }
    }
  }

  // Store f32 output [b][c][n]; written once -> non-temporal (keep L2 for V/P).
#pragma unroll
  for (int t = 0; t < 2; ++t) {
#pragma unroll
    for (int nt = 0; nt < 4; ++nt) {
      const int c0 = wave * 32 + t * 16 + half * 8;
      float* ob = out + ((size_t)b * CC + c0) * NN + n0 + nt * 16 + l15;
#pragma unroll
      for (int j = 0; j < 8; ++j)
        __builtin_nontemporal_store(acc[t * 4 + nt][j], &ob[(size_t)j * NN]);
    }
  }
}

// ---------------------------------------------------------------------------
// Launch. Workspace layout:
//   QT:  [B][N][32] f16  2 MB     KT: [B][N][32] f16  2 MB
//   Vt:  [B][32 ctiles][128 msteps][512] f16 (A-operand tiles) 32 MB
//   rowmax: [B][N] f32 128 KB     rowrcp: [B][N] f32 128 KB
// ---------------------------------------------------------------------------
extern "C" void kernel_launch(void* const* d_in, const int* in_sizes, int n_in,
                              void* d_out, int out_size, void* d_ws, size_t ws_size,
                              hipStream_t stream) {
  const float* x  = (const float*)d_in[0];
  const float* Wq = (const float*)d_in[1];
  const float* Wk = (const float*)d_in[2];
  const float* Wv = (const float*)d_in[3];
  float* out = (float*)d_out;

  char* ws = (char*)d_ws;
  _Float16* QT = (_Float16*)(ws);
  _Float16* KT = (_Float16*)(ws + (size_t)2 * 1024 * 1024);
  _Float16* Vt = (_Float16*)(ws + (size_t)4 * 1024 * 1024);
  float* rowmax = (float*)(ws + (size_t)36 * 1024 * 1024);
  float* rowrcp = (float*)(ws + (size_t)36 * 1024 * 1024 + 128 * 1024);

  // Q, K: 32 rows -> 2 row-tiles per wave, transposed [n][d] store.
  proj_kernel<2, 1><<<dim3(NN / 16, 1, BN), 32, 0, stream>>>(Wq, x, QT, DD);
  proj_kernel<2, 1><<<dim3(NN / 16, 1, BN), 32, 0, stream>>>(Wk, x, KT, DD);
  // V: 512 rows -> 8 row-tiles per wave (x B-tile reused 8x), tiled store.
  proj_kernel<8, 0><<<dim3(NN / 16, CC / 128, BN), 32, 0, stream>>>(Wv, x, Vt, CC);

  stats_kernel<<<dim3(NN / 64, BN), 32, 0, stream>>>(QT, KT, rowmax, rowrcp);
  attn_out_kernel<<<dim3(NN / 64, BN), 512, 0, stream>>>(QT, KT, Vt, rowmax,
                                                         rowrcp, out);
}
